// RefactorNeighborhoodAttention1D_43954695307308
// MI455X (gfx1250) — matrix operand order repaired and checked by reading, not compiled, not run on hardware
//
#include <hip/hip_runtime.h>
#include <cstdint>
#include <cstddef>

// Problem constants (B=1)
#define LSEQ   384
#define CCH    256
#define NH     8
#define HDIM   32
#define LLTOT  (LSEQ * LSEQ)   // 147456
#define SQRT_HD 5.656854249492381f

typedef __attribute__((ext_vector_type(16))) __bf16 v16bf;
typedef __attribute__((ext_vector_type(8)))  float  v8f;

union FragAB { uint4 u[2]; v16bf v; };

__device__ __forceinline__ unsigned short f2bf(float f) {
    unsigned u = __float_as_uint(f);
    u += 0x7FFFu + ((u >> 16) & 1u);          // round-to-nearest-even
    return (unsigned short)(u >> 16);
}
__device__ __forceinline__ float bf2f(unsigned short b) {
    return __uint_as_float(((unsigned)b) << 16);
}
__device__ __forceinline__ unsigned packbf(float a, float b) {
    return (unsigned)f2bf(a) | ((unsigned)f2bf(b) << 16);
}

// ---------------------------------------------------------------------------
// Pre-swizzle a 256x256 row-major weight W (out[c] = sum_k A[k]*W[c,k]) into
// the CDNA5 WMMA B-fragment layout (bf16):
//   frag[kt][ct][lane][i] = W[16*ct + (lane&15)][32*kt + 8*(lane>>4) + (i<8 ? i : i+8)]
// so a lane's 16-element B fragment is 32 contiguous bytes.
// ---------------------------------------------------------------------------
__global__ void swizzle_w(const float* __restrict__ W, unsigned short* __restrict__ out) {
    int id   = blockIdx.x * 256 + threadIdx.x;   // 0..4095
    int kt   = id >> 9;
    int ct   = (id >> 5) & 15;
    int lane = id & 31;
    int c     = ct * 16 + (lane & 15);
    int kbase = kt * 32 + (lane >> 4) * 8;
    const float* src = W + (size_t)c * 256 + kbase;
    unsigned short* dst = out + (size_t)id * 16;
#pragma unroll
    for (int i = 0; i < 16; ++i) dst[i] = f2bf(src[i < 8 ? i : i + 8]);
}

// ---------------------------------------------------------------------------
// Generic Y[r,c] = sum_k A[r,k]*W[c,k] (+bias), K=N=256, 32 rows per block.
// 256 threads = 8 waves; wave -> 2 column tiles x both rowblks (B-frag reuse x2).
// ---------------------------------------------------------------------------
__global__ void gemm256_bf16(const float* __restrict__ A,
                             const unsigned short* __restrict__ Wswz,
                             const float* __restrict__ bias,
                             float* __restrict__ Y) {
    extern __shared__ char smem_raw[];
    unsigned short* a_lds = (unsigned short*)smem_raw;       // 32x256 bf16 = 16KB
    const int t  = threadIdx.x;
    const int p0 = blockIdx.x * 32;

    { // stage A tile -> bf16 LDS
        int r = t >> 3, cb = (t & 7) * 32;
        const float4* src = (const float4*)(A + (size_t)(p0 + r) * 256 + cb);
        unsigned short* dst = a_lds + r * 256 + cb;
#pragma unroll
        for (int q = 0; q < 8; ++q) {
            float4 f = src[q];
            dst[q*4+0] = f2bf(f.x); dst[q*4+1] = f2bf(f.y);
            dst[q*4+2] = f2bf(f.z); dst[q*4+3] = f2bf(f.w);
        }
    }
    __syncthreads();

    const int lane = t & 31, wave = t >> 5;
    const int laneLo = lane & 15, laneHi = lane >> 4;
    const int ctb = wave * 2;                       // 2 column tiles per wave

    v8f acc[4];                                     // [rowblk*2 + q]
#pragma unroll
    for (int q = 0; q < 4; ++q) acc[q] = (v8f){0.f,0.f,0.f,0.f,0.f,0.f,0.f,0.f};

#pragma unroll
    for (int kt = 0; kt < 8; ++kt) {
        const int k0 = kt * 32;
        FragAB af[2];   // both rowblks; ISA 16-bit A layout (2 contiguous chunks)
#pragma unroll
        for (int rb = 0; rb < 2; ++rb) {
            const unsigned short* arow = a_lds + (rb * 16 + laneLo) * 256;
            af[rb].u[0] = *(const uint4*)(arow + k0 + 8 * laneHi);
            af[rb].u[1] = *(const uint4*)(arow + k0 + 16 + 8 * laneHi);
        }
#pragma unroll
        for (int q = 0; q < 2; ++q) {
            const int ct = ctb + q;
            FragAB bf;
            const uint4* bp = (const uint4*)(Wswz + ((size_t)((kt * 16 + ct) * 32 + lane)) * 16);
            bf.u[0] = bp[0]; bf.u[1] = bp[1];
            acc[0*2+q] = __builtin_amdgcn_wmma_f32_16x16x32_bf16(
                false, af[0].v, false, bf.v, (short)0, acc[0*2+q], false, false);
            acc[1*2+q] = __builtin_amdgcn_wmma_f32_16x16x32_bf16(
                false, af[1].v, false, bf.v, (short)0, acc[1*2+q], false, false);
        }
    }
#pragma unroll
    for (int rb = 0; rb < 2; ++rb)
#pragma unroll
    for (int q = 0; q < 2; ++q) {
        const int col = (ctb + q) * 16 + laneLo;
        const float bv = bias ? bias[col] : 0.f;
#pragma unroll
        for (int r = 0; r < 8; ++r) {
            int row = p0 + rb * 16 + laneHi * 8 + r;
            Y[(size_t)row * 256 + col] = acc[rb*2+q][r] + bv;
        }
    }
}

// ---------------------------------------------------------------------------
// Fused: LayerNorm(rpe row) -> dual GEMM (rpe_k, rpe_v) -> epilogue:
//   d[h,p]  = (q . (k + rpe_k_head)) * sqrt(hd)      (f32)
//   v1[h,p] = v + rpe_v_head                          (bf16, halves HBM traffic)
// 32 rows (p = i*384+j) per block; 512 WMMAs / block; each B fragment reused
// across both rowblks from registers (halves L2 weight traffic).
// ---------------------------------------------------------------------------
__global__ void rpe_fused(const float* __restrict__ rpe,
                          const float* __restrict__ gamma, const float* __restrict__ beta,
                          const unsigned short* __restrict__ WkSwz,
                          const unsigned short* __restrict__ WvSwz,
                          const float* __restrict__ bk, const float* __restrict__ bv,
                          const float* __restrict__ qp, const float* __restrict__ kp,
                          const float* __restrict__ vp,
                          float* __restrict__ d_ws, unsigned short* __restrict__ v1_ws) {
    extern __shared__ char smem_raw[];
    unsigned short* a_lds = (unsigned short*)smem_raw;        // 16KB
    float* rk_lds = (float*)(smem_raw + 16384);               // 32KB
    float* rv_lds = rk_lds + 32 * 256;                        // 32KB
    const int t = threadIdx.x;
    const int lane = t & 31, wave = t >> 5;
    const int p0 = blockIdx.x * 32;

    if (t == 0 && p0 + 32 < LLTOT)  // speculative prefetch of next block's rpe tile
        __builtin_prefetch(rpe + (size_t)(p0 + 32) * 256, 0, 1);

    // ---- LayerNorm stage: wave handles rows wave*4 .. wave*4+3 ----
#pragma unroll
    for (int rr = 0; rr < 4; ++rr) {
        const int r = wave * 4 + rr;
        const float* row = rpe + (size_t)(p0 + r) * 256;
        const float4* rp = (const float4*)row + lane * 2;
        float4 x0 = rp[0], x1 = rp[1];
        float s  = x0.x + x0.y + x0.z + x0.w + x1.x + x1.y + x1.z + x1.w;
        float s2 = x0.x*x0.x + x0.y*x0.y + x0.z*x0.z + x0.w*x0.w
                 + x1.x*x1.x + x1.y*x1.y + x1.z*x1.z + x1.w*x1.w;
#pragma unroll
        for (int m = 16; m >= 1; m >>= 1) {
            s  += __shfl_xor(s,  m, 32);
            s2 += __shfl_xor(s2, m, 32);
        }
        float mean = s * (1.f / 256.f);
        float var  = s2 * (1.f / 256.f) - mean * mean;   // biased variance
        float rstd = rsqrtf(var + 1e-5f);
        const int cb = lane * 8;
        unsigned short* dst = a_lds + r * 256 + cb;
        float vals[8] = {x0.x, x0.y, x0.z, x0.w, x1.x, x1.y, x1.z, x1.w};
#pragma unroll
        for (int e = 0; e < 8; ++e) {
            float nv = (vals[e] - mean) * rstd * gamma[cb + e] + beta[cb + e];
            dst[e] = f2bf(nv);
        }
    }
    __syncthreads();

    // ---- dual GEMM: wave -> (matrix, 4 column tiles) x both rowblks ----
    const int laneLo = lane & 15, laneHi = lane >> 4;
    const int mtx = wave >> 2;             // 0 -> rpe_k, 1 -> rpe_v
    const int ctb = (wave & 3) * 4;        // 4 column tiles per wave
    const unsigned short* Wswz = mtx ? WvSwz : WkSwz;

    v8f acc[8];                            // [rowblk*4 + q]
#pragma unroll
    for (int q = 0; q < 8; ++q) acc[q] = (v8f){0.f,0.f,0.f,0.f,0.f,0.f,0.f,0.f};

#pragma unroll
    for (int kt = 0; kt < 8; ++kt) {
        const int k0 = kt * 32;
        FragAB af[2];
#pragma unroll
        for (int rb = 0; rb < 2; ++rb) {
            const unsigned short* arow = a_lds + (rb * 16 + laneLo) * 256;
            af[rb].u[0] = *(const uint4*)(arow + k0 + 8 * laneHi);
            af[rb].u[1] = *(const uint4*)(arow + k0 + 16 + 8 * laneHi);
        }
#pragma unroll
        for (int q = 0; q < 4; ++q) {
            const int ct = ctb + q;
            FragAB bf;
            const uint4* bp = (const uint4*)(Wswz + ((size_t)((kt * 16 + ct) * 32 + lane)) * 16);
            bf.u[0] = bp[0]; bf.u[1] = bp[1];
            acc[0*4+q] = __builtin_amdgcn_wmma_f32_16x16x32_bf16(
                false, af[0].v, false, bf.v, (short)0, acc[0*4+q], false, false);
            acc[1*4+q] = __builtin_amdgcn_wmma_f32_16x16x32_bf16(
                false, af[1].v, false, bf.v, (short)0, acc[1*4+q], false, false);
        }
    }

    { // scatter fragments into f32 LDS tiles (+bias)
        float* tile = mtx ? rv_lds : rk_lds;
        const float* bb = mtx ? bv : bk;
#pragma unroll
        for (int rb = 0; rb < 2; ++rb)
#pragma unroll
        for (int q = 0; q < 4; ++q) {
            const int col = (ctb + q) * 16 + laneLo;
            const float bvv = bb[col];
#pragma unroll
            for (int r = 0; r < 8; ++r)
                tile[(rb * 16 + laneHi * 8 + r) * 256 + col] = acc[rb*4+q][r] + bvv;
        }
    }
    __syncthreads();

    // ---- epilogue: thread -> (row r, head h) ----
    {
        const int r = t >> 3, h = t & 7;
        const int p = p0 + r;
        const int i = p / LSEQ;
        // q[b,h,i,:] == qp[h*48 + i/8, 32*(i%8) : +32]  (reshape-without-transpose)
        const int qoff = (h * 48 + (i >> 3)) * 256 + 32 * (i & 7);
        const float* q  = qp + qoff;
        const float* k  = kp + qoff;
        const float* v  = vp + qoff;
        const float* rk = rk_lds + r * 256 + h * 32;
        const float* rv = rv_lds + r * 256 + h * 32;
        float dsum = 0.f;
#pragma unroll
        for (int dd = 0; dd < 32; ++dd) dsum += q[dd] * (k[dd] + rk[dd]);
        d_ws[(size_t)h * LLTOT + p] = dsum * SQRT_HD;   // reference divides by hd^-0.5

        // v1 in bf16: 64B per (h,p) row
        uint4* vo = (uint4*)(v1_ws + ((size_t)h * LLTOT + p) * 32);
#pragma unroll
        for (int g = 0; g < 4; ++g) {
            const int dd = g * 8;
            uint4 o;
            o.x = packbf(v[dd+0] + rv[dd+0], v[dd+1] + rv[dd+1]);
            o.y = packbf(v[dd+2] + rv[dd+2], v[dd+3] + rv[dd+3]);
            o.z = packbf(v[dd+4] + rv[dd+4], v[dd+5] + rv[dd+5]);
            o.w = packbf(v[dd+6] + rv[dd+6], v[dd+7] + rv[dd+7]);
            vo[g] = o;
        }
    }
}

// ---------------------------------------------------------------------------
// Windowed softmax over flattened L*L axis + weighted v1 sum + row-sum over j.
// One block per (i, h); lanes carry hd, waves stride j. The padded d window
// for the whole row (398 values) is staged once in LDS.
// ---------------------------------------------------------------------------
__global__ void softmax_qv(const float* __restrict__ d_ws,
                           const unsigned short* __restrict__ v1_ws,
                           float* __restrict__ O) {
    const int i    = blockIdx.x;   // 0..383
    const int h    = blockIdx.y;   // 0..7
    const int lane = threadIdx.x & 31;
    const int wave = threadIdx.x >> 5;
    __shared__ float dsh[LSEQ + 14];   // d_pad[i*384-7 .. i*384+390]
    __shared__ float red[8][32];

    const float* dh = d_ws + (size_t)h * LLTOT;
    const unsigned short* vh = v1_ws + (size_t)h * LLTOT * 32;

    for (int x = threadIdx.x; x < LSEQ + 14; x += 256) {
        int idx = i * LSEQ - 7 + x;
        dsh[x] = (idx >= 0 && idx < LLTOT) ? dh[idx] : 0.f;   // zero-pad logits
    }
    __syncthreads();

    float acc = 0.f;
    for (int j = wave; j < LSEQ; j += 8) {
        const int p = i * LSEQ + j;
        float lg[15];
        float m = -1e30f;
#pragma unroll
        for (int w = 0; w < 15; ++w) {
            float x = dsh[j + w];
            lg[w] = x;
            m = fmaxf(m, x);
        }
        float ssum = 0.f;
#pragma unroll
        for (int w = 0; w < 15; ++w) { lg[w] = __expf(lg[w] - m); ssum += lg[w]; }
        float inv = 1.f / ssum;
#pragma unroll
        for (int w = 0; w < 15; ++w) {
            int idx = p + w - 7;
            if (idx >= 0 && idx < LLTOT)
                acc += lg[w] * inv * bf2f(vh[(size_t)idx * 32 + lane]);
        }
    }
    red[wave][lane] = acc;
    __syncthreads();
    if (threadIdx.x < 32) {
        float s = 0.f;
#pragma unroll
        for (int w = 0; w < 8; ++w) s += red[w][threadIdx.x];
        O[(size_t)i * 256 + h * 32 + threadIdx.x] = s;
    }
}

// ---------------------------------------------------------------------------
extern "C" void kernel_launch(void* const* d_in, const int* in_sizes, int n_in,
                              void* d_out, int out_size, void* d_ws, size_t ws_size,
                              hipStream_t stream) {
    const float* Q       = (const float*)d_in[0];
    const float* K       = (const float*)d_in[1];
    const float* V       = (const float*)d_in[2];
    const float* rpe     = (const float*)d_in[3];
    const float* w_q     = (const float*)d_in[4];
    const float* w_k     = (const float*)d_in[5];
    const float* w_v     = (const float*)d_in[6];
    const float* ln_g    = (const float*)d_in[7];
    const float* ln_b    = (const float*)d_in[8];
    const float* W_rpe_k = (const float*)d_in[9];
    const float* b_rpe_k = (const float*)d_in[10];
    const float* W_rpe_v = (const float*)d_in[11];
    const float* b_rpe_v = (const float*)d_in[12];
    const float* W_proj  = (const float*)d_in[13];
    const float* b_proj  = (const float*)d_in[14];

    char* ws = (char*)d_ws;
    unsigned short* swz[6];
    for (int m = 0; m < 6; ++m) swz[m] = (unsigned short*)(ws + (size_t)m * 131072);
    float* qp    = (float*)(ws + 6 * 131072);
    float* kp    = qp + LSEQ * CCH;
    float* vp    = kp + LSEQ * CCH;
    float* Obuf  = vp + LSEQ * CCH;
    float* d_buf = Obuf + LSEQ * CCH;                         // 8 * LL floats
    unsigned short* v1 = (unsigned short*)(d_buf + (size_t)NH * LLTOT);  // 8*LL*32 bf16

    const float* wsrc[6] = { w_q, w_k, w_v, W_rpe_k, W_rpe_v, W_proj };
    for (int m = 0; m < 6; ++m)
        swizzle_w<<<16, 256, 0, stream>>>(wsrc[m], swz[m]);

    gemm256_bf16<<<LSEQ / 32, 256, 16384, stream>>>(Q, swz[0], nullptr, qp);
    gemm256_bf16<<<LSEQ / 32, 256, 16384, stream>>>(K, swz[1], nullptr, kp);
    gemm256_bf16<<<LSEQ / 32, 256, 16384, stream>>>(V, swz[2], nullptr, vp);

    rpe_fused<<<LLTOT / 32, 256, 81920, stream>>>(rpe, ln_g, ln_b, swz[3], swz[4],
                                                  b_rpe_k, b_rpe_v, qp, kp, vp,
                                                  d_buf, v1);

    softmax_qv<<<dim3(LSEQ, NH), 256, 0, stream>>>(d_buf, v1, Obuf);

    gemm256_bf16<<<LSEQ / 32, 256, 16384, stream>>>(Obuf, swz[5], b_proj, (float*)d_out);

    (void)in_sizes; (void)n_in; (void)out_size; (void)ws_size;
}
